// EdgeConv_32942399160848
// MI455X (gfx1250) — matrix operand order repaired and checked by reading, not compiled, not run on hardware
//
#include <hip/hip_runtime.h>
#include <stdint.h>

// EdgeConv (DGCNN) for MI455X / gfx1250, wave32 + WMMA + async-to-LDS staging.
// B=8, C=64, N=4096, K=20, O=64, edge channels = 128.

#define Bn 8
#define Cn 64
#define Nn 4096
#define Kn 20
#define On 64

typedef __attribute__((ext_vector_type(16))) _Float16 v16h;
typedef __attribute__((ext_vector_type(8)))  _Float16 h8;
typedef __attribute__((ext_vector_type(8)))  float    v8f;
typedef __attribute__((ext_vector_type(4)))  float    f4;
typedef __attribute__((ext_vector_type(4)))  int      v4i;

// ---- CDNA5 async global->LDS copy (ASYNCcnt), with guarded fallback --------
#if defined(__has_builtin)
#if __has_builtin(__builtin_amdgcn_global_load_async_to_lds_b128)
#define HAVE_ASYNC_LDS 1
#endif
#endif

__device__ __forceinline__ void copy16_g2l(_Float16* l, const _Float16* g) {
#ifdef HAVE_ASYNC_LDS
  typedef __attribute__((address_space(1))) v4i* g4p;
  typedef __attribute__((address_space(3))) v4i* l4p;
  __builtin_amdgcn_global_load_async_to_lds_b128(
      (g4p)(uintptr_t)g, (l4p)(uint32_t)(uintptr_t)l, 0, 0);
#else
  *(h8*)l = *(const h8*)g;
#endif
}

#ifdef HAVE_ASYNC_LDS
#if __has_builtin(__builtin_amdgcn_s_wait_asynccnt)
#define WAIT_ASYNC(n) __builtin_amdgcn_s_wait_asynccnt(n)
#else
#define WAIT_ASYNC(n) asm volatile("s_wait_asynccnt %0" ::"i"(n))
#endif
#else
#define WAIT_ASYNC(n) ((void)0)
#endif

// ---------------------------------------------------------------------------
// Kernel 1: xh16[b][n][c] (f16, n-major) and xx[b][n] = sum_c x^2.
// ---------------------------------------------------------------------------
__global__ void __launch_bounds__(256) prep_kernel(const float* __restrict__ x,
                                                   _Float16* __restrict__ xh,
                                                   float* __restrict__ xx) {
  int b  = blockIdx.x >> 6;
  int n0 = (blockIdx.x & 63) * 64;
  int tid = threadIdx.x;
  int nn = tid & 63, c4 = tid >> 6;
  const float* xb = x + (size_t)b * Cn * Nn;
  _Float16* xhb = xh + (size_t)b * Nn * Cn;
  float s = 0.f;
#pragma unroll
  for (int cc = 0; cc < 16; ++cc) {
    int c = c4 * 16 + cc;
    float v = xb[(size_t)c * Nn + n0 + nn];
    s += v * v;
    xhb[(size_t)(n0 + nn) * Cn + c] = (_Float16)v;
  }
  __shared__ float red[256];
  red[tid] = s;
  __syncthreads();
  if (tid < 64)
    xx[b * Nn + n0 + tid] = red[tid] + red[64 + tid] + red[128 + tid] + red[192 + tid];
}

// ---------------------------------------------------------------------------
// Kernel 2: kNN. Block = 256 thr = 8 waves, each wave owns a 16-query tile
// (128 queries/block). Candidate rows streamed once per block through LDS
// (double-buffered async copies, 128 rows/chunk, rows padded to 72 halves).
// Each lane: 1 query, 2048-candidate subset, private sorted top-20 via
// branch-guarded shift insert; 2-list merge per query at the end.
// ---------------------------------------------------------------------------
__global__ void __launch_bounds__(256) knn_kernel(const float* __restrict__ x,
                                                  const _Float16* __restrict__ xh,
                                                  const float* __restrict__ xx,
                                                  int* __restrict__ idxout) {
  __shared__ __align__(16) _Float16 stage[2][128 * 72];  // 36 KB
  __shared__ float mval[8 * 16 * 2 * Kn];                // 20 KB
  __shared__ int   midx[8 * 16 * 2 * Kn];                // 20 KB

  int b    = blockIdx.x >> 5;    // 32 query-blocks per batch
  int qblk = blockIdx.x & 31;
  int tid  = threadIdx.x;
  int lane = tid & 31, wave = tid >> 5;
  int qcol = lane & 15, hi = lane >> 4;
  int q = qblk * 128 + wave * 16 + qcol;

  const float*    xb  = x  + (size_t)b * Cn * Nn;
  const _Float16* xhb = xh + (size_t)b * Nn * Cn;
  const float*    xxb = xx + b * Nn;

  // B fragment (queries, 32x16 per k-step): half h -> K = 8*hi + (h&7) + 16*(h>>3) (+32*ks)
  v16h bf0, bf1;
#pragma unroll
  for (int h = 0; h < 16; ++h) {
    int c = hi * 8 + (h & 7) + (h >> 3) * 16;
    bf0[h] = (_Float16)xb[(size_t)c * Nn + q];
    bf1[h] = (_Float16)xb[(size_t)(c + 32) * Nn + q];
  }
  float xxq = xxb[q];

  float bv[Kn]; int bi[Kn];
#pragma unroll
  for (int k = 0; k < Kn; ++k) { bv[k] = -3.0e38f; bi[k] = 0; }

  // stage chunk ch (128 rows) into buffer buf: 256 thr x 4 x 16B pieces
  auto stage_chunk = [&](int ch, int buf) {
    const _Float16* gsrc = xhb + (size_t)ch * 128 * Cn;
#pragma unroll
    for (int p = 0; p < 4; ++p) {
      int cidx = p * 256 + tid;            // 0..1023 16B pieces
      int row = cidx >> 3, piece = cidx & 7;
      copy16_g2l(&stage[buf][row * 72 + piece * 8], gsrc + row * 64 + piece * 8);
    }
  };

  stage_chunk(0, 0);
  for (int ch = 0; ch < 32; ++ch) {
    if (ch < 31) {
      stage_chunk(ch + 1, (ch + 1) & 1);
      WAIT_ASYNC(4);                       // current chunk's 4 ops retired
    } else {
      WAIT_ASYNC(0);
    }
    __syncthreads();                       // all waves' copies visible

    const _Float16* sbase = stage[ch & 1];
#pragma unroll 2
    for (int t = 0; t < 8; ++t) {
      int mbase = ch * 128 + t * 16;
      // A fragment (candidates 16x32): lane row = qcol; halves per ISA layout
      const _Float16* rowp = sbase + (t * 16 + qcol) * 72 + hi * 8;
      h8 t0 = *(const h8*)(rowp);
      h8 t1 = *(const h8*)(rowp + 16);
      h8 t2 = *(const h8*)(rowp + 32);
      h8 t3 = *(const h8*)(rowp + 48);
      v16h a0, a1;
#pragma unroll
      for (int i = 0; i < 8; ++i) {
        a0[i] = t0[i]; a0[8 + i] = t1[i];
        a1[i] = t2[i]; a1[8 + i] = t3[i];
      }
      v8f acc = {};
      acc = __builtin_amdgcn_wmma_f32_16x16x32_f16(false, a0, false, bf0, (short)0, acc, false, false);
      acc = __builtin_amdgcn_wmma_f32_16x16x32_f16(false, a1, false, bf1, (short)0, acc, false, false);

      f4 xa = *(const f4*)(xxb + mbase + hi * 8);
      f4 xc = *(const f4*)(xxb + mbase + hi * 8 + 4);
      float xxm[8];
#pragma unroll
      for (int i = 0; i < 4; ++i) { xxm[i] = xa[i]; xxm[4 + i] = xc[i]; }

#pragma unroll
      for (int r = 0; r < 8; ++r) {
        int   m  = mbase + hi * 8 + r;     // D row M = r + 8*hi
        float pd = 2.f * acc[r] - xxq - xxm[r];
        if (pd > bv[Kn - 1]) {             // shift-style sorted insert
          bool gt[Kn];
#pragma unroll
          for (int j = 0; j < Kn; ++j) gt[j] = pd > bv[j];
#pragma unroll
          for (int j = Kn - 1; j >= 1; --j) {
            bv[j] = gt[j] ? (gt[j - 1] ? bv[j - 1] : pd) : bv[j];
            bi[j] = gt[j] ? (gt[j - 1] ? bi[j - 1] : m)  : bi[j];
          }
          bv[0] = gt[0] ? pd : bv[0];
          bi[0] = gt[0] ? m  : bi[0];
        }
      }
    }
    __syncthreads();                       // done reading before overwrite
  }

  // Merge two per-lane lists per query.
  int slot = ((wave * 16 + qcol) * 2 + hi) * Kn;
#pragma unroll
  for (int k = 0; k < Kn; ++k) { mval[slot + k] = bv[k]; midx[slot + k] = bi[k]; }
  __syncthreads();

  if (tid < 128) {
    int w2 = tid >> 4, qc = tid & 15;
    int qg = qblk * 128 + w2 * 16 + qc;
    const float* va = mval + ((w2 * 16 + qc) * 2 + 0) * Kn;
    const float* vb = mval + ((w2 * 16 + qc) * 2 + 1) * Kn;
    const int*   ia = midx + ((w2 * 16 + qc) * 2 + 0) * Kn;
    const int*   ib = midx + ((w2 * 16 + qc) * 2 + 1) * Kn;
    int* op = idxout + ((size_t)b * Nn + qg) * Kn;
    int i = 0, j = 0;
    for (int k = 0; k < Kn; ++k) {
      int ii = i < Kn ? i : Kn - 1;
      int jj = j < Kn ? j : Kn - 1;
      float av = va[ii], bw = vb[jj];
      int   ai = ia[ii], bj = ib[jj];
      bool ta = (j >= Kn) || (i < Kn && (av > bw || (av == bw && ai < bj)));
      op[k] = ta ? ai : bj;
      if (ta) ++i; else ++j;
    }
  }
}

// ---------------------------------------------------------------------------
// Kernel 3: P[b][m][o] = x_row(m).W1[o], Q[b][n][o] = x_row(n).(W2[o]-W1[o]).
// ---------------------------------------------------------------------------
__global__ void __launch_bounds__(256) proj_kernel(const float* __restrict__ x,
                                                   const float* __restrict__ W,
                                                   float* __restrict__ P,
                                                   float* __restrict__ Q) {
  __shared__ float sW[On * 129];
  __shared__ float sx[4 * Cn];
  int b  = blockIdx.x >> 10;
  int m0 = (blockIdx.x & 1023) * 4;
  int tid = threadIdx.x;
  for (int i = tid; i < On * 128; i += 256) {
    int o = i >> 7, c = i & 127;
    sW[o * 129 + c] = W[i];
  }
  {
    int mm = tid & 3, c = tid >> 2;
    sx[mm * Cn + c] = x[((size_t)b * Cn + c) * Nn + m0 + mm];
  }
  __syncthreads();
  int mm = tid >> 6, o = tid & 63;
  const float* xr = sx + mm * Cn;
  const float* wr = sW + o * 129;
  float p = 0.f, qv = 0.f;
#pragma unroll
  for (int c = 0; c < Cn; ++c) {
    float xc = xr[c], w1 = wr[c], w2 = wr[64 + c];
    p  += xc * w1;
    qv += xc * (w2 - w1);
  }
  size_t oo = ((size_t)b * Nn + m0 + mm) * On + o;
  P[oo] = p; Q[oo] = qv;
}

// ---------------------------------------------------------------------------
// Kernel 4: per (b,o): gather y = P[idx]+Q, instance-norm stats over (n,k),
// then normalize + leaky-relu + max over k -> out[b][o][n].
// ---------------------------------------------------------------------------
__global__ void __launch_bounds__(256) normout_kernel(const float* __restrict__ P,
                                                      const float* __restrict__ Q,
                                                      const int* __restrict__ idx,
                                                      float* __restrict__ out) {
  int b = blockIdx.x >> 6, o = blockIdx.x & 63;
  int tid = threadIdx.x;
  const float* Pb = P + (size_t)b * Nn * On + o;
  const float* Qb = Q + (size_t)b * Nn * On + o;
  const int*   ib = idx + (size_t)b * Nn * Kn;

  float s1 = 0.f, s2 = 0.f;
  for (int n = tid; n < Nn; n += 256) {
    float qv = Qb[(size_t)n * On];
    const int* ip = ib + n * Kn;
#pragma unroll
    for (int k = 0; k < Kn; ++k) {
      float y = Pb[(size_t)ip[k] * On] + qv;
      s1 += y; s2 += y * y;
    }
  }
  __shared__ float r1[256], r2[256];
  r1[tid] = s1; r2[tid] = s2;
  __syncthreads();
  for (int s = 128; s > 0; s >>= 1) {
    if (tid < s) { r1[tid] += r1[tid + s]; r2[tid] += r2[tid + s]; }
    __syncthreads();
  }
  __shared__ float mean_s, rstd_s;
  if (tid == 0) {
    const float inv = 1.f / (float)(Nn * Kn);
    float mean = r1[0] * inv;
    float var  = r2[0] * inv - mean * mean;
    mean_s = mean;
    rstd_s = rsqrtf(var + 1e-5f);
  }
  __syncthreads();
  float mean = mean_s, rstd = rstd_s;

  float* ob = out + ((size_t)b * On + o) * Nn;
  for (int n = tid; n < Nn; n += 256) {
    float qv = Qb[(size_t)n * On];
    const int* ip = ib + n * Kn;
    float mx = -3.4e38f;
#pragma unroll
    for (int k = 0; k < Kn; ++k) {
      float y = (Pb[(size_t)ip[k] * On] + qv - mean) * rstd;
      y = y > 0.f ? y : 0.2f * y;
      mx = fmaxf(mx, y);
    }
    ob[n] = mx;
  }
}

// ---------------------------------------------------------------------------
// Launcher. Workspace (bytes):
//   xh16 [0,4194304)  xx [4194304,4325376)  idx [4325376,6946816)
//   P [6946816,15335424)  Q [15335424,23724032)
// ---------------------------------------------------------------------------
extern "C" void kernel_launch(void* const* d_in, const int* in_sizes, int n_in,
                              void* d_out, int out_size, void* d_ws, size_t ws_size,
                              hipStream_t stream) {
  const float* x = (const float*)d_in[0];
  const float* W = (const float*)d_in[1];
  float* out = (float*)d_out;
  char* ws = (char*)d_ws;

  _Float16* xh = (_Float16*)(ws);
  float*    xx = (float*)(ws + 4194304);
  int*      idx = (int*)(ws + 4325376);
  float*    P  = (float*)(ws + 6946816);
  float*    Q  = (float*)(ws + 15335424);

  prep_kernel<<<Bn * (Nn / 64), 256, 0, stream>>>(x, xh, xx);
  knn_kernel<<<Bn * (Nn / 128), 256, 0, stream>>>(x, xh, xx, idx);
  proj_kernel<<<Bn * (Nn / 4), 256, 0, stream>>>(x, W, P, Q);
  normout_kernel<<<Bn * On, 256, 0, stream>>>(P, Q, idx, out);
}
